// Social_Aggregator_31069793419779
// MI455X (gfx1250) — matrix operand order repaired and checked by reading, not compiled, not run on hardware
//
#include <hip/hip_runtime.h>

#define NNODES 4096
#define DEG    64
#define D      256

typedef __attribute__((ext_vector_type(16))) __bf16 v16bf;
typedef __attribute__((ext_vector_type(8)))  float  v8f;

union FragU { uint4 q[2]; v16bf v; };

__device__ __forceinline__ __bf16 f2bf(float f) {
  unsigned u = __builtin_bit_cast(unsigned, f);
  u += 0x7FFFu + ((u >> 16) & 1u);            // round-to-nearest-even
  unsigned short h = (unsigned short)(u >> 16);
  return __builtin_bit_cast(__bf16, h);
}
__device__ __forceinline__ float bf2f(__bf16 b) {
  unsigned short h = __builtin_bit_cast(unsigned short, b);
  unsigned u = ((unsigned)h) << 16;
  return __builtin_bit_cast(float, u);
}

__device__ __forceinline__ v8f wmma_bf16(v16bf a, v16bf b, v8f c) {
  // (neg_a, A, neg_b, B, c_mod, C, reuse_a, reuse_b)
  return __builtin_amdgcn_wmma_f32_16x16x32_bf16(false, a, false, b, (short)0, c,
                                                 false, false);
}

// ---------------------------------------------------------------------------
// Shared GEMM core: ks-outer so each B fragment (global, L2-resident) is
// loaded ONCE and reused by all 4 M-tiles (8 WMMAs per B-fragment pair).
// A-fragments come from LDS; k-steps >= NKA read the M-invariant ego
// embedding (the [X ; u] concat), which the compiler hoists to 1 ds_load
// pair feeding 8 WMMAs.
// ---------------------------------------------------------------------------
template <int NKA, int NKTOT>
__device__ __forceinline__ void gemm_tiles(const __bf16* __restrict__ Abuf,
                                           const __bf16* __restrict__ Upart,
                                           const __bf16* __restrict__ W,
                                           int nt0, int nt1, int lane, int ml,
                                           int hiA, v8f acc[4][2]) {
#pragma unroll
  for (int ks = 0; ks < NKTOT; ++ks) {
    const uint4* bp0 = (const uint4*)(W + (((nt0 * NKTOT + ks) << 9) + lane * 16));
    const uint4* bp1 = (const uint4*)(W + (((nt1 * NKTOT + ks) << 9) + lane * 16));
    FragU fb0; fb0.q[0] = bp0[0]; fb0.q[1] = bp0[1];
    FragU fb1; fb1.q[0] = bp1[0]; fb1.q[1] = bp1[1];
#pragma unroll
    for (int mt = 0; mt < 4; ++mt) {
      const __bf16* ap = (ks < NKA)
                             ? (Abuf + (mt * 16 + ml) * D + ks * 32 + hiA * 2)
                             : (Upart + (ks - NKA) * 32 + hiA * 2);
      FragU fa; fa.q[0] = *(const uint4*)ap; fa.q[1] = *(const uint4*)(ap + 8);
      acc[mt][0] = wmma_bf16(fa.v, fb0.v, acc[mt][0]);
      acc[mt][1] = wmma_bf16(fa.v, fb1.v, acc[mt][1]);
    }
  }
}

// ---------------------------------------------------------------------------
// Weight pre-pass: fp32 row-major [K x 256] -> bf16 tiles in WMMA-B fragment
// order. Fragment for (ntile, kstep): 32 lanes x 16 bf16 contiguous.
// B order used here (32x16, K contiguous): lane 0-15 -> N=lane,    K=kstep*32+e
//                           lane 16-31 -> N=lane-16, K=kstep*32+16+e
// ---------------------------------------------------------------------------
__global__ void conv_w(const float* __restrict__ src, __bf16* __restrict__ dst,
                       int nK, int N, int total) {
  int idx = blockIdx.x * blockDim.x + threadIdx.x;
  if (idx >= total) return;
  int tile   = idx >> 9;        // 512 halfs per 32x16 tile
  int within = idx & 511;
  int lane   = within >> 4;
  int e      = within & 15;
  int ntile  = tile / nK;
  int kstep  = tile - ntile * nK;
  int k = kstep * 32 + ((lane & 16) ? 16 : 0) + e;
  int n = ntile * 16 + (lane & 15);
  dst[idx] = f2bf(src[k * N + n]);
}

// ---------------------------------------------------------------------------
// Main kernel: one block (8 waves, 256 threads) per node.
// ---------------------------------------------------------------------------
__global__ void __launch_bounds__(256)
social_agg(const int* __restrict__ nodes, const int* __restrict__ neighs,
           const float* __restrict__ u2e,
           const __bf16* __restrict__ Wg, const float* __restrict__ bg,
           const __bf16* __restrict__ W1, const float* __restrict__ b1,
           const __bf16* __restrict__ W2, const float* __restrict__ b2,
           const float* __restrict__ W3, const float* __restrict__ b3,
           float* __restrict__ out) {
  __shared__ __align__(16) __bf16 Xbuf[DEG * D];   // e_u, then e_g (32 KB)
  __shared__ __align__(16) __bf16 Hbuf[DEG * D];   // h1 (32 KB)
  __shared__ __align__(16) __bf16 Ulds[D];         // ego embedding (bf16)
  __shared__ float Sc[DEG];
  __shared__ float Att[DEG];

  const int nid  = blockIdx.x;
  const int t    = threadIdx.x;
  const int lane = t & 31;
  const int wv   = t >> 5;

  // ---- Phase 0: gather ego + 64 neighbor embeddings, convert to bf16 ----
  const long unode = (long)nodes[nid];
  Ulds[t] = f2bf(u2e[unode * D + t]);
  if (t < DEG) Sc[t] = 0.f;
  {
    const int k   = t >> 2;          // 4 threads per neighbor row
    const int seg = (t & 3) * 64;
    const long base = (long)neighs[nid * DEG + k] * D + seg;
    __bf16* dst = Xbuf + k * D + seg;
    for (int i = 0; i < 64; i += 4) {
      float4 v = *(const float4*)(u2e + base + i);
      union { __bf16 h[4]; uint2 u; } pk;
      pk.h[0] = f2bf(v.x); pk.h[1] = f2bf(v.y);
      pk.h[2] = f2bf(v.z); pk.h[3] = f2bf(v.w);
      *(uint2*)(dst + i) = pk.u;
    }
  }
  __syncthreads();

  const int hiA = (lane & 16) ? 8 : 0;   // A-frag K offset for high lanes
  const int ml  = lane & 15;
  const int nt0 = wv * 2, nt1 = wv * 2 + 1;   // each wave owns 2 N-tiles

  const v8f vzero = {0.f, 0.f, 0.f, 0.f, 0.f, 0.f, 0.f, 0.f};
  v8f accs[4][2];

  // ---- Phase 1: gate GEMM  [64 x 512] x [512 x 256], K-split: X | u ----
#pragma unroll
  for (int mt = 0; mt < 4; ++mt) { accs[mt][0] = vzero; accs[mt][1] = vzero; }
  gemm_tiles<8, 16>(Xbuf, Ulds, Wg, nt0, nt1, lane, ml, hiA, accs);
  __syncthreads();   // everyone done reading e_u before gating overwrites it

  // ---- Phase 2: gating  e_g = sig(z)*e_u + (1-sig(z))*u  (in place) ----
  for (int mt = 0; mt < 4; ++mt) {
    for (int j = 0; j < 2; ++j) {
      const int n  = (nt0 + j) * 16 + ml;
      const float gbv = bg[n];
      const float uu  = bf2f(Ulds[n]);
      v8f a = accs[mt][j];
      for (int r = 0; r < 8; ++r) {
        const int m = mt * 16 + r + hiA;
        const float gv = 1.f / (1.f + __expf(-(a[r] + gbv)));
        const float eu = bf2f(Xbuf[m * D + n]);
        Xbuf[m * D + n] = f2bf(gv * eu + (1.f - gv) * uu);
      }
    }
  }
  __syncthreads();   // e_g ready

  // ---- Phase 3: att1 GEMM  relu([e_g ; u] @ W1 + b1) -> Hbuf ----
#pragma unroll
  for (int mt = 0; mt < 4; ++mt) { accs[mt][0] = vzero; accs[mt][1] = vzero; }
  gemm_tiles<8, 16>(Xbuf, Ulds, W1, nt0, nt1, lane, ml, hiA, accs);
  for (int mt = 0; mt < 4; ++mt) {
    for (int j = 0; j < 2; ++j) {
      const int n  = (nt0 + j) * 16 + ml;
      const float bb = b1[n];
      v8f a = accs[mt][j];
      for (int r = 0; r < 8; ++r) {
        const int m = mt * 16 + r + hiA;
        float h = a[r] + bb;
        Hbuf[m * D + n] = f2bf(h > 0.f ? h : 0.f);
      }
    }
  }
  __syncthreads();   // h1 ready

  // ---- Phase 4: att2 GEMM + fused score = relu(.)·att3_W, reduce -> Sc ----
#pragma unroll
  for (int mt = 0; mt < 4; ++mt) { accs[mt][0] = vzero; accs[mt][1] = vzero; }
  gemm_tiles<8, 8>(Hbuf, Ulds, W2, nt0, nt1, lane, ml, hiA, accs);
  for (int mt = 0; mt < 4; ++mt) {
    for (int j = 0; j < 2; ++j) {
      const int n  = (nt0 + j) * 16 + ml;
      const float bb = b2[n];
      const float w3 = W3[n];
      v8f a = accs[mt][j];
      for (int r = 0; r < 8; ++r) {
        float h = a[r] + bb;
        float v = (h > 0.f ? h : 0.f) * w3;
        // reduce over the 16 N-lanes of this half-wave
        for (int off = 1; off < 16; off <<= 1) v += __shfl_xor(v, off, 32);
        if (ml == 0) atomicAdd(&Sc[mt * 16 + r + hiA], v);
      }
    }
  }
  __syncthreads();

  // ---- Phase 5: softmax over the 64 neighbors ----
  if (t == 0) {
    const float b3v = b3[0];
    float mx = -3.4e38f;
    for (int k = 0; k < DEG; ++k) { float s = Sc[k] + b3v; mx = s > mx ? s : mx; }
    float sum = 0.f;
    for (int k = 0; k < DEG; ++k) { float e = __expf(Sc[k] + b3v - mx); Att[k] = e; sum += e; }
    const float inv = 1.f / sum;
    for (int k = 0; k < DEG; ++k) Att[k] *= inv;
  }
  __syncthreads();

  // ---- Phase 6: out[d] = sum_k Att[k] * e_g[k][d] ----
  float acc = 0.f;
  for (int k = 0; k < DEG; ++k) acc += Att[k] * bf2f(Xbuf[k * D + t]);
  out[(long)nid * D + t] = acc;
}

// ---------------------------------------------------------------------------
extern "C" void kernel_launch(void* const* d_in, const int* in_sizes, int n_in,
                              void* d_out, int out_size, void* d_ws, size_t ws_size,
                              hipStream_t stream) {
  const int*   nodes  = (const int*)d_in[0];
  const int*   neighs = (const int*)d_in[1];
  const float* u2e    = (const float*)d_in[2];
  const float* gW     = (const float*)d_in[3];
  const float* gb     = (const float*)d_in[4];
  const float* a1W    = (const float*)d_in[5];
  const float* a1b    = (const float*)d_in[6];
  const float* a2W    = (const float*)d_in[7];
  const float* a2b    = (const float*)d_in[8];
  const float* a3W    = (const float*)d_in[9];
  const float* a3b    = (const float*)d_in[10];

  __bf16* Wg = (__bf16*)d_ws;                 // 512x256 bf16 = 256 KB
  __bf16* W1 = Wg + 512 * 256;                // 512x256 bf16 = 256 KB
  __bf16* W2 = W1 + 512 * 256;                // 256x256 bf16 = 128 KB

  conv_w<<<512, 256, 0, stream>>>(gW,  Wg, 16, 256, 512 * 256);
  conv_w<<<512, 256, 0, stream>>>(a1W, W1, 16, 256, 512 * 256);
  conv_w<<<256, 256, 0, stream>>>(a2W, W2,  8, 256, 256 * 256);

  social_agg<<<NNODES, 256, 0, stream>>>(nodes, neighs, u2e,
                                         Wg, gb, W1, a1b, W2, a2b, a3W, a3b,
                                         (float*)d_out);
}
